// BitLinear_74620761800729
// MI455X (gfx1250) — 8-bit matrix operand order repaired and checked by reading, compiled, not run on hardware
//
#include <hip/hip_runtime.h>


typedef int v8i __attribute__((ext_vector_type(8)));

constexpr int   BATCH = 4;
constexpr int   SEQ   = 2048;
constexpr int   M     = BATCH * SEQ;
constexpr int   Kdim  = 4096;
constexpr int   Nout  = 16384;
constexpr float EPS   = 1e-5f;

constexpr int MB    = 128;
constexpr int NB    = 256;
constexpr int KB    = 64;
constexpr int LDST  = 80;
constexpr int ASZ   = MB * LDST;
constexpr int BSZ   = NB * LDST;
constexpr int BUFSZ = ASZ + BSZ;

constexpr size_t XQ_OFF    = 0;
constexpr size_t WQ_OFF    = XQ_OFF + (size_t)M * Kdim;
constexpr size_t INV_OFF   = WQ_OFF + (size_t)Nout * Kdim;
constexpr size_t PART_OFF  = INV_OFF + (size_t)M * sizeof(float);
constexpr size_t SCALE_OFF = PART_OFF + 1024 * sizeof(float);

#if __has_builtin(__builtin_amdgcn_global_load_async_to_lds_b128) && \
    __has_builtin(__builtin_amdgcn_s_wait_asynccnt)
#define USE_ASYNC_LDS 1
#endif

#ifdef USE_ASYNC_LDS
typedef int vs4i __attribute__((vector_size(16)));
typedef __attribute__((address_space(1))) vs4i glb_vs4i;
typedef __attribute__((address_space(3))) vs4i lds_vs4i;

__device__ __forceinline__ glb_vs4i* to_gbl(const void* p) {
    return (glb_vs4i*)(unsigned long long)(__SIZE_TYPE__)p;
}
__device__ __forceinline__ lds_vs4i* to_lds(const void* p) {
    return (lds_vs4i*)(unsigned int)(unsigned long long)(__SIZE_TYPE__)p;
}
#endif

__device__ __forceinline__ void join_barrier() {
#ifdef USE_ASYNC_LDS
    __builtin_amdgcn_s_wait_asynccnt(0);
#endif
    __syncthreads();
}

__device__ __forceinline__ void stage_tile(const signed char* aG, const signed char* bG,
                                           char* aL, char* bL) {
#ifdef USE_ASYNC_LDS
    __builtin_amdgcn_global_load_async_to_lds_b128(to_gbl(aG +  0), to_lds(aL +  0), 0, 0);
    __builtin_amdgcn_global_load_async_to_lds_b128(to_gbl(aG + 16), to_lds(aL + 16), 0, 0);
    __builtin_amdgcn_global_load_async_to_lds_b128(to_gbl(bG +  0), to_lds(bL +  0), 0, 0);
    __builtin_amdgcn_global_load_async_to_lds_b128(to_gbl(bG + 16), to_lds(bL + 16), 0, 0);
    __builtin_amdgcn_global_load_async_to_lds_b128(to_gbl(bG + 32), to_lds(bL + 32), 0, 0);
    __builtin_amdgcn_global_load_async_to_lds_b128(to_gbl(bG + 48), to_lds(bL + 48), 0, 0);
#else
    int4 a0 = ((const int4*)aG)[0];
    int4 a1 = ((const int4*)aG)[1];
    int4 b0 = ((const int4*)bG)[0];
    int4 b1 = ((const int4*)bG)[1];
    int4 b2 = ((const int4*)bG)[2];
    int4 b3 = ((const int4*)bG)[3];
    ((int4*)aL)[0] = a0;
    ((int4*)aL)[1] = a1;
    ((int4*)bL)[0] = b0;
    ((int4*)bL)[1] = b1;
    ((int4*)bL)[2] = b2;
    ((int4*)bL)[3] = b3;
#endif
}

__global__ void wabs_sum_part(const float* __restrict__ w, float* __restrict__ part) {
    __shared__ float sdata[256];
    const int tid = threadIdx.x;
    const size_t n4     = (size_t)Nout * Kdim / 4;
    const size_t stride = (size_t)gridDim.x * blockDim.x;
    const float4* w4 = (const float4*)w;
    float s = 0.0f;
    for (size_t j = (size_t)blockIdx.x * blockDim.x + tid; j < n4; j += stride) {
        float4 v = w4[j];
        s += fabsf(v.x) + fabsf(v.y) + fabsf(v.z) + fabsf(v.w);
    }
    sdata[tid] = s;
    __syncthreads();
    for (int k = 128; k > 0; k >>= 1) {
        if (tid < k) sdata[tid] += sdata[tid + k];
        __syncthreads();
    }
    if (tid == 0) part[blockIdx.x] = sdata[0];
}

__global__ void wscale_finalize(const float* __restrict__ part, float* __restrict__ scale2) {
    __shared__ float sdata[256];
    const int tid = threadIdx.x;
    sdata[tid] = part[tid] + part[tid + 256] + part[tid + 512] + part[tid + 768];
    __syncthreads();
    for (int k = 128; k > 0; k >>= 1) {
        if (tid < k) sdata[tid] += sdata[tid + k];
        __syncthreads();
    }
    if (tid == 0) {
        float mean  = sdata[0] * (1.0f / ((float)Nout * (float)Kdim));
        float meanc = fmaxf(mean, EPS);
        scale2[0] = 1.0f / meanc;
        scale2[1] = meanc;
    }
}

__global__ void wquant(const float* __restrict__ w, const float* __restrict__ scale2,
                       signed char* __restrict__ wq) {
    const float qs = scale2[0];
    const size_t t = (size_t)blockIdx.x * blockDim.x + threadIdx.x;
    const float4* w4 = (const float4*)w + t * 4;
    union { signed char c[16]; int4 v; } u;
#pragma unroll
    for (int j = 0; j < 4; ++j) {
        float4 v = w4[j];
        u.c[4 * j + 0] = (signed char)(int)fminf(fmaxf(rintf(v.x * qs), -1.0f), 1.0f);
        u.c[4 * j + 1] = (signed char)(int)fminf(fmaxf(rintf(v.y * qs), -1.0f), 1.0f);
        u.c[4 * j + 2] = (signed char)(int)fminf(fmaxf(rintf(v.z * qs), -1.0f), 1.0f);
        u.c[4 * j + 3] = (signed char)(int)fminf(fmaxf(rintf(v.w * qs), -1.0f), 1.0f);
    }
    ((int4*)wq)[t] = u.v;
}

__global__ void aquant(const float* __restrict__ x, signed char* __restrict__ xq,
                       float* __restrict__ inv_scale) {
    __shared__ float sdata[256];
    const int row = blockIdx.x;
    const int tid = threadIdx.x;
    const float4* xr4 = (const float4*)(x + (size_t)row * Kdim);
    float4 v[4];
    float am = 0.0f;
#pragma unroll
    for (int j = 0; j < 4; ++j) {
        v[j] = xr4[tid * 4 + j];
        am = fmaxf(am, fmaxf(fmaxf(fabsf(v[j].x), fabsf(v[j].y)),
                             fmaxf(fabsf(v[j].z), fabsf(v[j].w))));
    }
    sdata[tid] = am;
    __syncthreads();
    for (int k = 128; k > 0; k >>= 1) {
        if (tid < k) sdata[tid] = fmaxf(sdata[tid], sdata[tid + k]);
        __syncthreads();
    }
    const float amc = fmaxf(sdata[0], EPS);
    const float qs  = 127.0f / amc;
    union { signed char c[16]; int4 q; } u;
#pragma unroll
    for (int j = 0; j < 4; ++j) {
        u.c[4 * j + 0] = (signed char)(int)fminf(fmaxf(rintf(v[j].x * qs), -128.0f), 127.0f);
        u.c[4 * j + 1] = (signed char)(int)fminf(fmaxf(rintf(v[j].y * qs), -128.0f), 127.0f);
        u.c[4 * j + 2] = (signed char)(int)fminf(fmaxf(rintf(v[j].z * qs), -128.0f), 127.0f);
        u.c[4 * j + 3] = (signed char)(int)fminf(fmaxf(rintf(v[j].w * qs), -128.0f), 127.0f);
    }
    ((int4*)(xq + (size_t)row * Kdim))[tid] = u.q;
    if (tid == 0) inv_scale[row] = amc * (1.0f / 127.0f);
}

__global__ __launch_bounds__(256)
void bitlinear_gemm(const signed char* __restrict__ xq,
                    const signed char* __restrict__ wq,
                    const float* __restrict__ inv_scale,
                    const float* __restrict__ scale2,
                    float* __restrict__ out) {
    __shared__ char lds[2 * BUFSZ];

    const int tid  = threadIdx.x;
    const int lane = tid & 31;
    const int wave = tid >> 5;
    const int wm   = wave & 1;
    const int wn   = wave >> 1;
    const int l15  = lane & 15;
    const int lhi  = lane >> 4;
    const int mblk = blockIdx.y * MB;
    const int nblk = blockIdx.x * NB;

    const signed char* aG = xq + (size_t)(mblk + (tid >> 1)) * Kdim + (tid & 1) * 32;
    const signed char* bG = wq + (size_t)(nblk + tid) * Kdim;
    char* aL = lds + (tid >> 1) * LDST + (tid & 1) * 32;
    char* bL = lds + ASZ + tid * LDST;

    const int aOffL = (wm * 64 + l15) * LDST + lhi * 16;
    const int bOffL = ASZ + (wn * 64 + l15) * LDST + lhi * 16;

    const v8i vzero = {0, 0, 0, 0, 0, 0, 0, 0};
    v8i acc[4][4];
#pragma unroll
    for (int i = 0; i < 4; ++i)
#pragma unroll
        for (int j = 0; j < 4; ++j) acc[i][j] = vzero;

    stage_tile(aG, bG, aL, bL);
    join_barrier();

    constexpr int NK = Kdim / KB;
    for (int it = 0; it < NK; ++it) {
        if (it + 1 < NK) {
            const int nb = (it + 1) & 1;
            stage_tile(aG + (it + 1) * KB, bG + (it + 1) * KB,
                       aL + nb * BUFSZ, bL + nb * BUFSZ);
        }

        const char* base = lds + (it & 1) * BUFSZ;

        v8i bfrag[4];
#pragma unroll
        for (int ni = 0; ni < 4; ++ni) {
            const char* p = base + bOffL + ni * 16 * LDST;
            int4 c0 = *(const int4*)(p + 0);
            int4 c1 = *(const int4*)(p + 32);
            bfrag[ni][0] = c0.x; bfrag[ni][1] = c0.y;
            bfrag[ni][2] = c0.z; bfrag[ni][3] = c0.w;
            bfrag[ni][4] = c1.x; bfrag[ni][5] = c1.y;
            bfrag[ni][6] = c1.z; bfrag[ni][7] = c1.w;
        }
#pragma unroll
        for (int mi = 0; mi < 4; ++mi) {
            const char* q = base + aOffL + mi * 16 * LDST;
            int2 t0 = *(const int2*)(q + 0);
            int2 t1 = *(const int2*)(q + 8);
            int2 t2 = *(const int2*)(q + 32);
            int2 t3 = *(const int2*)(q + 40);
            v8i afrag;
            afrag[0] = t0.x; afrag[1] = t0.y;
            afrag[2] = t1.x; afrag[3] = t1.y;
            afrag[4] = t2.x; afrag[5] = t2.y;
            afrag[6] = t3.x; afrag[7] = t3.y;
#pragma unroll
            for (int ni = 0; ni < 4; ++ni)
                acc[mi][ni] = __builtin_amdgcn_wmma_i32_16x16x64_iu8(
                     true, afrag,  true, bfrag[ni],
                    acc[mi][ni],  false,  false);
        }

        join_barrier();
    }

    const float wsc = scale2[1];
#pragma unroll
    for (int mi = 0; mi < 4; ++mi) {
        const int mbase = mblk + wm * 64 + mi * 16 + lhi * 8;
        float isc[8];
#pragma unroll
        for (int r = 0; r < 8; ++r) isc[r] = inv_scale[mbase + r] * wsc;
#pragma unroll
        for (int ni = 0; ni < 4; ++ni) {
            const int n = nblk + wn * 64 + ni * 16 + l15;
#pragma unroll
            for (int r = 0; r < 8; ++r)
                out[(size_t)(mbase + r) * Nout + n] = (float)acc[mi][ni][r] * isc[r];
        }
    }
}

extern "C" void kernel_launch(void* const* d_in, const int* in_sizes, int n_in,
                              void* d_out, int out_size, void* d_ws, size_t ws_size,
                              hipStream_t stream) {
    const float* x = (const float*)d_in[0];
    const float* w = (const float*)d_in[1];
    float* out = (float*)d_out;

    char* ws = (char*)d_ws;
    signed char* xq  = (signed char*)(ws + XQ_OFF);
    signed char* wq  = (signed char*)(ws + WQ_OFF);
    float* inv_scale = (float*)(ws + INV_OFF);
    float* part      = (float*)(ws + PART_OFF);
    float* scale2    = (float*)(ws + SCALE_OFF);

    wabs_sum_part<<<1024, 256, 0, stream>>>(w, part);
    wscale_finalize<<<1, 256, 0, stream>>>(part, scale2);

    wquant<<<(Nout * (size_t)Kdim) / (16 * 256), 256, 0, stream>>>(w, scale2, wq);

    aquant<<<M, 256, 0, stream>>>(x, xq, inv_scale);

    dim3 grid(Nout / NB, M / MB);
    bitlinear_gemm<<<grid, 256, 0, stream>>>(xq, wq, inv_scale, scale2, out);
}
